// Decoder_58308476010663
// MI455X (gfx1250) — matrix operand order repaired and checked by reading, not compiled, not run on hardware
//
#include <hip/hip_runtime.h>

// ---------------------------------------------------------------------------
// Autoregressive LSTM decoder for MI455X (gfx1250).
//   BATCH=64, LSTM_DIM=2048, MEL=512, INPUT=512, STEPS=512.
//
// Key algebraic fusion: x_t = y_{t-1} = h_{t-1} @ W_out^T + b_out, so
//   gates_t = h_{t-1} @ (W_hh + W_ih@W_out)^T + (b_ih+b_hh + W_ih@b_out)
// for t>=1 (t=0 uses plain W_hh since x_0 = 0). This removes the x-GEMM from
// the critical path and needs only ONE grid barrier per step (publish h_t).
// y_{t-1} is computed by 32 dedicated blocks concurrently with the gate GEMM.
//
// bf16 weights/activations, fp32 WMMA accumulation, fp32 cell state. All
// weights (~76MB bf16 incl. W_comb) stay L2-resident (192MB L2); the 64MB
// output is written with nontemporal stores to avoid polluting L2.
// ---------------------------------------------------------------------------

typedef __attribute__((ext_vector_type(16))) __bf16 v16bf;
typedef __attribute__((ext_vector_type(8)))  float  v8f;

#define LSTM_H   2048
#define XDIM     512
#define MELD     512
#define BATCH    64
#define STEPS    512
#define NBLOCKS  160   // 128 gate blocks + 32 output blocks

// ---- workspace layout (bytes) ----
#define OFF_WCOMB ((size_t)0)          // 8192*2048 bf16 = 33554432
#define OFF_WHH   ((size_t)33554432)   // 8192*2048 bf16 = 33554432
#define OFF_WIH   ((size_t)67108864)   // 8192*512  bf16 =  8388608
#define OFF_WOUT  ((size_t)75497472)   // 512*2048  bf16 =  2097152
#define OFF_WT    ((size_t)77594624)   // 2048*512  bf16 =  2097152 (W_out^T)
#define OFF_BIAS  ((size_t)79691776)   // 8192 f32       =    32768
#define OFF_BIASC ((size_t)79724544)   // 8192 f32       =    32768
#define OFF_C     ((size_t)79757312)   // 64*2048 f32    =   524288
#define OFF_HB0   ((size_t)80281600)   // 64*2048 bf16   =   262144
#define OFF_HB1   ((size_t)80543744)   // 64*2048 bf16   =   262144
#define OFF_CNT   ((size_t)80805888)   // u32 barrier counter

__device__ __forceinline__ unsigned short f2bf(float f) {
  unsigned u = __float_as_uint(f);
  unsigned r = (u + 0x7FFFu + ((u >> 16) & 1u)) >> 16;  // round-nearest-even
  return (unsigned short)r;
}

union V16 { uint4 q[2]; v16bf v; };

// A tile: 16x32 bf16 (M x K) from row-major src (ld halfs per row).
// ISA layout: lanes0-15: M=lane, halfs0-7 = K0..7, halfs8-15 = K16..23;
//             lanes16-31: M=lane-16, halfs0-7 = K8..15, halfs8-15 = K24..31.
__device__ __forceinline__ v16bf load_a_tile(const unsigned short* src,
                                             int ld, int m0, int k0) {
  int lane = threadIdx.x & 31;
  int m    = m0 + (lane & 15);
  int khi  = (lane >> 4) * 8;
  const unsigned short* p = src + (size_t)m * ld + k0 + khi;
  V16 t;
  t.q[0] = *(const uint4*)(p);        // K = k0+khi   .. +7
  t.q[1] = *(const uint4*)(p + 16);   // K = k0+khi+16.. +23
  return t.v;
}

// B tile: 32x16 bf16 (K x N); computing act @ W^T means B(k,n) = W[n_glob][k],
// so each B column is a contiguous row of W (ld halfs).
// ISA layout: N = lane&15; lanes0-15 hold K=k0..+7,+16..+23; lanes16-31 K=+8..+15,+24..+31.
__device__ __forceinline__ v16bf load_b_tile(const unsigned short* w,
                                             int ld, int n0, int k0) {
  int lane = threadIdx.x & 31;
  int n    = n0 + (lane & 15);
  int koff = (lane >> 4) * 8;
  const unsigned short* p = w + (size_t)n * ld + k0 + koff;
  V16 t;
  t.q[0] = *(const uint4*)(p);
  t.q[1] = *(const uint4*)(p + 16);
  return t.v;
}

__device__ __forceinline__ v8f wmma_bf16(v16bf a, v16bf b, v8f c) {
  return __builtin_amdgcn_wmma_f32_16x16x32_bf16(false, a, false, b,
                                                 (short)0, c, false, false);
}

__device__ __forceinline__ float sigmoidf_(float x) {
  return 1.0f / (1.0f + __expf(-x));
}
__device__ __forceinline__ float tanhf_(float x) {
  float e = __expf(2.0f * x);          // saturates correctly at +/-inf
  return 1.0f - 2.0f / (e + 1.0f);
}

// Grid-wide barrier (all NBLOCKS co-resident; 0 LDS, 4 waves/block).
__device__ __forceinline__ void grid_barrier(unsigned* cnt, unsigned target) {
  __threadfence();          // release: flush this wave's stores to device scope
  __syncthreads();
  if (threadIdx.x == 0) {
    __hip_atomic_fetch_add(cnt, 1u, __ATOMIC_RELEASE, __HIP_MEMORY_SCOPE_AGENT);
    while (__hip_atomic_load(cnt, __ATOMIC_ACQUIRE, __HIP_MEMORY_SCOPE_AGENT) < target) {
      __builtin_amdgcn_s_sleep(1);
    }
  }
  __syncthreads();
  __threadfence();          // acquire: invalidate so fresh cross-block reads
}

// ------------------------- prologue kernels -------------------------------
__global__ void k_f2bf(unsigned short* dst, const float* src, int n) {
  int i = blockIdx.x * blockDim.x + threadIdx.x;
  if (i < n) dst[i] = f2bf(src[i]);
}
__global__ void k_copy(float* dst, const float* src, int n) {
  int i = blockIdx.x * blockDim.x + threadIdx.x;
  if (i < n) dst[i] = src[i];
}
// W_out^T (bf16): Wt[k][j] = Wout[j][k];  Wout is (512, 2048) f32.
__global__ void k_tr(unsigned short* wt, const float* wout) {
  int i = blockIdx.x * blockDim.x + threadIdx.x;   // i = k*512 + j
  if (i < MELD * LSTM_H) {
    int k = i >> 9, j = i & 511;
    wt[i] = f2bf(wout[(size_t)j * LSTM_H + k]);
  }
}
// bias = b_ih+b_hh ; bias_c = bias + W_ih @ b_out ; also reset barrier cnt.
__global__ void k_biasc(float* bias, float* bias_c, const float* bih,
                        const float* bhh, const float* wih_f,
                        const float* bout, unsigned* cnt) {
  int o = blockIdx.x * blockDim.x + threadIdx.x;
  if (o == 0) *cnt = 0;
  if (o < 4 * LSTM_H) {
    float s = bih[o] + bhh[o];
    bias[o] = s;
    const float* row = wih_f + (size_t)o * XDIM;
    float d = 0.f;
    for (int j = 0; j < XDIM; ++j) d = fmaf(row[j], bout[j], d);
    bias_c[o] = s + d;
  }
}
// W_comb = bf16( Whh_f32 + Wih_bf16 @ Wt_bf16 )   [8192 x 2048], K = 512.
// Grid: 4096 blocks x 128 thr; block = (m-tile, n-group); wave does 4 N-tiles.
__global__ void __launch_bounds__(128)
k_fuse(const unsigned short* Wih, const unsigned short* Wt,
       const float* Whh_f, unsigned short* Wcomb) {
  const int m0   = (blockIdx.x & 511) * 16;
  const int wv   = threadIdx.x >> 5;
  const int nb   = (blockIdx.x >> 9) * 256 + wv * 64;
  const int lane = threadIdx.x & 31;
  v8f acc[4] = {{}, {}, {}, {}};
#pragma unroll 4
  for (int k0 = 0; k0 < XDIM; k0 += 32) {
    v16bf a = load_a_tile(Wih, XDIM, m0, k0);
#pragma unroll
    for (int i = 0; i < 4; ++i)
      acc[i] = wmma_bf16(a, load_b_tile(Wt, XDIM, nb + i * 16, k0), acc[i]);
  }
  const int nL = lane & 15;
  const int mb = m0 + ((lane >> 4) * 8);
#pragma unroll
  for (int i = 0; i < 4; ++i)
#pragma unroll
    for (int r = 0; r < 8; ++r) {
      size_t idx = (size_t)(mb + r) * LSTM_H + (nb + i * 16 + nL);
      Wcomb[idx] = f2bf(Whh_f[idx] + acc[i][r]);
    }
}

// ------------------------- persistent LSTM kernel -------------------------
__global__ void __launch_bounds__(128)
lstm_persistent(const unsigned short* __restrict__ Wcomb,
                const unsigned short* __restrict__ Whh,
                const unsigned short* __restrict__ Wout,
                const float* __restrict__ bias,
                const float* __restrict__ bias_c,
                const float* __restrict__ b_out,
                float* __restrict__ c,
                unsigned short* hb0, unsigned short* hb1,
                float* __restrict__ out,
                unsigned* cnt) {
  const int bid  = blockIdx.x;
  const int wv   = threadIdx.x >> 5;
  const int lane = threadIdx.x & 31;
  const int m0   = wv * 16;                     // 16 batch rows per wave
  const int nL   = lane & 15;
  const int mb   = m0 + ((lane >> 4) * 8);      // M base for C/D rows
  const bool is_gate = (bid < LSTM_H / 16);     // blocks 0..127
  const int j0 = bid * 16;                      // gate blocks: hidden cols
  const int n0 = (bid - LSTM_H / 16) * 16;      // y blocks: mel cols

  unsigned short* hcur  = hb0;
  unsigned short* hnext = hb1;
  unsigned gen = 0;

#pragma unroll 1
  for (int t = 0; t <= STEPS; ++t) {
    if (is_gate) {
      if (t < STEPS) {
        // gates_t = h_{t-1} @ Wg^T + bv   (one K=2048 GEMM; fused for t>=1)
        const unsigned short* Wg = (t == 0) ? Whh : Wcomb;
        const float* bv = (t == 0) ? bias : bias_c;
        v8f acc_i = {}, acc_f = {}, acc_g = {}, acc_o = {};
#pragma unroll 4
        for (int k0 = 0; k0 < LSTM_H; k0 += 32) {
          v16bf a = load_a_tile(hcur, LSTM_H, m0, k0);
          acc_i = wmma_bf16(a, load_b_tile(Wg, LSTM_H, 0 * LSTM_H + j0, k0), acc_i);
          acc_f = wmma_bf16(a, load_b_tile(Wg, LSTM_H, 1 * LSTM_H + j0, k0), acc_f);
          acc_g = wmma_bf16(a, load_b_tile(Wg, LSTM_H, 2 * LSTM_H + j0, k0), acc_g);
          acc_o = wmma_bf16(a, load_b_tile(Wg, LSTM_H, 3 * LSTM_H + j0, k0), acc_o);
        }
        const int jcol = j0 + nL;
        const float bi = bv[0 * LSTM_H + jcol];
        const float bf = bv[1 * LSTM_H + jcol];
        const float bg = bv[2 * LSTM_H + jcol];
        const float bo = bv[3 * LSTM_H + jcol];
#pragma unroll
        for (int r = 0; r < 8; ++r) {
          const size_t idx = (size_t)(mb + r) * LSTM_H + jcol;
          float cv = c[idx];
          float gi = sigmoidf_(acc_i[r] + bi);
          float gf = sigmoidf_(acc_f[r] + bf);
          float gg = tanhf_(acc_g[r] + bg);
          float go = sigmoidf_(acc_o[r] + bo);
          float cn = gf * cv + gi * gg;
          c[idx] = cn;                           // fp32 cell state, in place
          hnext[idx] = f2bf(go * tanhf_(cn));    // bf16 h for next matmuls
        }
      }
    } else if (t > 0) {
      // y_{t-1} = h_{t-1} @ W_out^T + b_out  (reads hcur; off critical path)
      v8f acc = {};
#pragma unroll 4
      for (int k0 = 0; k0 < LSTM_H; k0 += 32) {
        v16bf a = load_a_tile(hcur, LSTM_H, m0, k0);
        acc = wmma_bf16(a, load_b_tile(Wout, LSTM_H, n0, k0), acc);
      }
      const int oc = n0 + nL;
      const float bb = b_out[oc];
#pragma unroll
      for (int r = 0; r < 8; ++r) {
        const int row = mb + r;
        float y = acc[r] + bb;                   // out[b][t-1][oc], streamed
        __builtin_nontemporal_store(
            y, &out[(size_t)row * (STEPS * MELD) + (size_t)(t - 1) * MELD + oc]);
      }
    }

    if (t < STEPS) {                             // publish h_t, then swap
      ++gen;
      grid_barrier(cnt, (unsigned)NBLOCKS * gen);
      unsigned short* tmp = hcur; hcur = hnext; hnext = tmp;
    }
  }
}

// ------------------------------ launch -------------------------------------
extern "C" void kernel_launch(void* const* d_in, const int* in_sizes, int n_in,
                              void* d_out, int out_size, void* d_ws, size_t ws_size,
                              hipStream_t stream) {
  // setup_inputs order:
  // 0 encoder_output (unused), 1 h0, 2 c0, 3 W_ih, 4 W_hh, 5 b_ih, 6 b_hh,
  // 7 W_out, 8 b_out
  const float* h0   = (const float*)d_in[1];
  const float* c0   = (const float*)d_in[2];
  const float* Wih  = (const float*)d_in[3];
  const float* Whh  = (const float*)d_in[4];
  const float* bih  = (const float*)d_in[5];
  const float* bhh  = (const float*)d_in[6];
  const float* Wout = (const float*)d_in[7];
  const float* bout = (const float*)d_in[8];
  float* out = (float*)d_out;

  char* ws = (char*)d_ws;
  unsigned short* Wcomb_b = (unsigned short*)(ws + OFF_WCOMB);
  unsigned short* Whh_b   = (unsigned short*)(ws + OFF_WHH);
  unsigned short* Wih_b   = (unsigned short*)(ws + OFF_WIH);
  unsigned short* Wout_b  = (unsigned short*)(ws + OFF_WOUT);
  unsigned short* Wt_b    = (unsigned short*)(ws + OFF_WT);
  float*          biasb   = (float*)(ws + OFF_BIAS);
  float*          biascb  = (float*)(ws + OFF_BIASC);
  float*          cbuf    = (float*)(ws + OFF_C);
  unsigned short* hb0     = (unsigned short*)(ws + OFF_HB0);
  unsigned short* hb1     = (unsigned short*)(ws + OFF_HB1);
  unsigned*       cnt     = (unsigned*)(ws + OFF_CNT);

  const int nWhh  = 4 * LSTM_H * LSTM_H;   // 16,777,216
  const int nWih  = 4 * LSTM_H * XDIM;     //  4,194,304
  const int nWout = MELD * LSTM_H;         //  1,048,576
  const int nH    = BATCH * LSTM_H;        //    131,072
  const int nBias = 4 * LSTM_H;            //      8,192

  k_f2bf<<<(nWhh  + 255) / 256, 256, 0, stream>>>(Whh_b,  Whh,  nWhh);
  k_f2bf<<<(nWih  + 255) / 256, 256, 0, stream>>>(Wih_b,  Wih,  nWih);
  k_f2bf<<<(nWout + 255) / 256, 256, 0, stream>>>(Wout_b, Wout, nWout);
  k_f2bf<<<(nH    + 255) / 256, 256, 0, stream>>>(hb0,    h0,   nH);   // h0[0]
  k_tr  <<<(nWout + 255) / 256, 256, 0, stream>>>(Wt_b, Wout);
  k_biasc<<<(nBias + 255) / 256, 256, 0, stream>>>(biasb, biascb, bih, bhh,
                                                   Wih, bout, cnt);
  k_copy<<<(nH    + 255) / 256, 256, 0, stream>>>(cbuf, c0, nH);       // c0[0]
  // W_comb = Whh + Wih @ Wout  (WMMA, fp32 Whh added before bf16 rounding)
  k_fuse<<<4096, 128, 0, stream>>>(Wih_b, Wt_b, Whh, Wcomb_b);

  lstm_persistent<<<NBLOCKS, 128, 0, stream>>>(Wcomb_b, Whh_b, Wout_b,
                                               biasb, biascb, bout,
                                               cbuf, hb0, hb1, out, cnt);
}
